// SelectiveSSM_15908558865281
// MI455X (gfx1250) — matrix operand order repaired and checked by reading, not compiled, not run on hardware
//
#include <hip/hip_runtime.h>

// ---------------- problem constants ----------------
#define D_MODEL   1024
#define D_STATE   16
#define D_CONV    4
#define D_INNER   2048
#define DT_RANK   64
#define B_SZ      2
#define L_SEQ     2048
#define N_TOK     (B_SZ * L_SEQ)          // 4096 tokens
#define XP_COLS   (DT_RANK + 2 * D_STATE) // 96

typedef __bf16 bf16_t;
typedef __attribute__((ext_vector_type(16))) __bf16 v16bf;
typedef __attribute__((ext_vector_type(8)))  float  v8f;

// ---------------- fp32 -> bf16 convert ----------------
__global__ void k_f32_to_bf16(const float* __restrict__ src, bf16_t* __restrict__ dst, int n) {
    int i = blockIdx.x * blockDim.x + threadIdx.x;
    if (i < n) dst[i] = (bf16_t)src[i];
}

// strided slice convert: dst[r*cols+c] = bf16(src[r*ld+col0+c])
__global__ void k_slice_to_bf16(const float* __restrict__ src, int ld, int col0, int cols,
                                bf16_t* __restrict__ dst, int rows) {
    int i = blockIdx.x * blockDim.x + threadIdx.x;
    if (i >= rows * cols) return;
    int r = i / cols, c = i - r * cols;
    dst[i] = (bf16_t)src[r * ld + col0 + c];
}

// ---------------- WMMA bf16 GEMM: C[M,N] = A[M,K] * W[N,K]^T ----------------
// A row-major [M,K] bf16, W row-major [N,K] bf16 (nn.Linear weight), C fp32 [M,N].
// One wave computes a 64x32 tile: 4 A-fragments x 2 B-fragments -> 8
// v_wmma_f32_16x16x32_bf16 per 32-deep k-step (8 WMMAs / 12 b128 loads),
// ~130 VGPRs so no accumulator spilling (64x64 spilled to scratch).

// A fragment (16x32): lane<16 holds row m, K = k0..k0+15;
//                     lane>=16 same rows, K = k0+16..k0+31.
// Also issues a zero-register-cost global_prefetch_b8 of the next k-step
// (offset folds into the instruction's immediate field).
__device__ __forceinline__ v16bf load_fragA(const bf16_t* __restrict__ A, int ld,
                                            int row, int k0, int lane) {
    const int khalf = (lane & 16) ? 16 : 0;
    const bf16_t* p0 = A + (size_t)row * ld + k0 + khalf;
    __builtin_prefetch(p0 + 32, 0, 1);   // next 32-deep k-step, speculative
    v16bf f;
#pragma unroll
    for (int i = 0; i < 16; ++i) f[i] = p0[i];
    return f;
}

// B fragment (32x16): lane<16 holds column n with K = k0..k0+15,
//                     lane>=16 same columns with K = k0+16..k0+31.
// B[k][n] == W[n][k], so this is a contiguous row-major read of W.
__device__ __forceinline__ v16bf load_fragB(const bf16_t* __restrict__ W, int ld,
                                            int col, int k0, int lane) {
    const int koff = (lane & 16) ? 16 : 0;
    const bf16_t* p = W + (size_t)col * ld + k0 + koff;
    __builtin_prefetch(p + 32, 0, 1);
    v16bf f;
#pragma unroll
    for (int i = 0; i < 16; ++i) f[i] = p[i];
    return f;
}

template <int MT, int NT>
__global__ void __launch_bounds__(32, 1)
k_gemm_bf16_wmma(const bf16_t* __restrict__ A, const bf16_t* __restrict__ W,
                 float* __restrict__ C, int M, int N, int K) {
    const int lane = threadIdx.x;              // blockDim.x == 32, one wave per block
    const int m0   = blockIdx.x * (MT * 16);
    const int n0   = blockIdx.y * (NT * 16);
    const int arow = lane & 15;

    v8f acc[MT][NT] = {};

    for (int k0 = 0; k0 < K; k0 += 32) {
        v16bf a[MT], b[NT];
#pragma unroll
        for (int i = 0; i < MT; ++i) a[i] = load_fragA(A, K, m0 + i * 16 + arow, k0, lane);
#pragma unroll
        for (int j = 0; j < NT; ++j) b[j] = load_fragB(W, K, n0 + j * 16 + arow, k0, lane);

#pragma unroll
        for (int i = 0; i < MT; ++i)
#pragma unroll
            for (int j = 0; j < NT; ++j)
                acc[i][j] = __builtin_amdgcn_wmma_f32_16x16x32_bf16(
                    false, a[i], false, b[j], (short)0, acc[i][j], false, false);
    }

    // C/D layout: lane<16 -> col=lane, rows r..; lane>=16 -> col=lane-16, rows 8+r.
    const int col   = lane & 15;
    const int rbase = (lane & 16) ? 8 : 0;
#pragma unroll
    for (int i = 0; i < MT; ++i)
#pragma unroll
        for (int j = 0; j < NT; ++j)
#pragma unroll
            for (int r = 0; r < 8; ++r)
                C[(size_t)(m0 + i * 16 + rbase + r) * N + n0 + j * 16 + col] = acc[i][j][r];
}

// ---------------- causal depthwise conv (k=4) + bias + SiLU ----------------
// xz: [N_TOK, 2*D_INNER] fp32 ; x_branch = cols [0,D_INNER)
__global__ void k_conv_silu(const float* __restrict__ xz,
                            const float* __restrict__ cw,    // [D_INNER,1,4]
                            const float* __restrict__ cb,    // [D_INNER]
                            float* __restrict__ xconv_f,
                            bf16_t* __restrict__ xconv_b) {
    int idx = blockIdx.x * blockDim.x + threadIdx.x;
    if (idx >= N_TOK * D_INNER) return;
    const int d = idx & (D_INNER - 1);
    const int t = idx >> 11;                 // token index (b*L + l)
    const int l = t & (L_SEQ - 1);
    float acc = cb[d];
#pragma unroll
    for (int j = 0; j < D_CONV; ++j) {
        const int ls = l - (D_CONV - 1) + j;
        if (ls >= 0)
            acc += cw[d * D_CONV + j] * xz[(size_t)(t - (D_CONV - 1) + j) * (2 * D_INNER) + d];
    }
    const float s = acc / (1.0f + __expf(-acc));   // SiLU
    xconv_f[idx] = s;
    xconv_b[idx] = (bf16_t)s;
}

// ---------------- bias + softplus (in place) ----------------
__global__ void k_bias_softplus(float* __restrict__ dt, const float* __restrict__ bias,
                                int rows, int cols) {
    int idx = blockIdx.x * blockDim.x + threadIdx.x;
    if (idx >= rows * cols) return;
    const float v = dt[idx] + bias[idx & (cols - 1)];
    dt[idx] = (v > 20.0f) ? v : log1pf(__expf(v));
}

// ---------------- selective scan (sequential over L) ----------------
// One thread owns one (b, d) channel; h[16] in registers; B_t/C_t staged in LDS.
// Fuses  y = (scan + D*x_conv) * silu(z)  and emits bf16 for out_proj.
__global__ void k_scan(const float* __restrict__ dt,      // [N_TOK, D_INNER]
                       const float* __restrict__ xconv,   // [N_TOK, D_INNER]
                       const float* __restrict__ xp,      // [N_TOK, 96]
                       const float* __restrict__ xz,      // [N_TOK, 2*D_INNER] (z = cols 2048+)
                       const float* __restrict__ A_log,   // [D_INNER, 16]
                       const float* __restrict__ Dp,      // [D_INNER]
                       bf16_t* __restrict__ ybf) {        // [N_TOK, D_INNER]
    const int tid = threadIdx.x;                           // 256 threads
    const int b   = blockIdx.x / (D_INNER / 256);
    const int d   = (blockIdx.x % (D_INNER / 256)) * 256 + tid;

    __shared__ float sB[D_STATE];
    __shared__ float sC[D_STATE];

    float Ad[D_STATE], h[D_STATE];
#pragma unroll
    for (int n = 0; n < D_STATE; ++n) {
        Ad[n] = -__expf(A_log[d * D_STATE + n]);
        h[n]  = 0.0f;
    }
    const float Dd = Dp[d];

    for (int l = 0; l < L_SEQ; ++l) {
        const int row = b * L_SEQ + l;
        if (tid < D_STATE)
            sB[tid] = xp[(size_t)row * XP_COLS + DT_RANK + tid];
        else if (tid < 2 * D_STATE)
            sC[tid - D_STATE] = xp[(size_t)row * XP_COLS + DT_RANK + D_STATE + (tid - D_STATE)];
        __syncthreads();

        const float dtv = dt[(size_t)row * D_INNER + d];
        const float xv  = xconv[(size_t)row * D_INNER + d];
        const float dbx = dtv * xv;
        float y = 0.0f;
#pragma unroll
        for (int n = 0; n < D_STATE; ++n) {
            h[n] = __expf(dtv * Ad[n]) * h[n] + dbx * sB[n];
            y += h[n] * sC[n];
        }
        const float zv = xz[(size_t)row * (2 * D_INNER) + D_INNER + d];
        const float out = (y + Dd * xv) * (zv / (1.0f + __expf(-zv)));
        ybf[(size_t)row * D_INNER + d] = (bf16_t)out;
        __syncthreads();
    }
}

// ---------------- launcher ----------------
static inline size_t align256(size_t x) { return (x + 255) & ~(size_t)255; }

extern "C" void kernel_launch(void* const* d_in, const int* in_sizes, int n_in,
                              void* d_out, int out_size, void* d_ws, size_t ws_size,
                              hipStream_t stream) {
    (void)in_sizes; (void)n_in; (void)out_size; (void)ws_size;

    const float* x         = (const float*)d_in[0];   // [B,L,1024]
    const float* in_proj_w = (const float*)d_in[1];   // [4096,1024]
    const float* conv_w    = (const float*)d_in[2];   // [2048,1,4]
    const float* conv_b    = (const float*)d_in[3];   // [2048]
    const float* x_proj_w  = (const float*)d_in[4];   // [96,2048]
    const float* dt_proj_w = (const float*)d_in[5];   // [2048,64]
    const float* dt_proj_b = (const float*)d_in[6];   // [2048]
    const float* A_log     = (const float*)d_in[7];   // [2048,16]
    const float* D_param   = (const float*)d_in[8];   // [2048]
    const float* out_proj_w= (const float*)d_in[9];   // [1024,2048]
    float* out = (float*)d_out;                        // [B,L,1024]

    // workspace carve-out
    char* ws = (char*)d_ws;
    size_t off = 0;
    auto alloc = [&](size_t bytes) { char* p = ws + off; off += align256(bytes); return p; };

    bf16_t* x_bf    = (bf16_t*)alloc((size_t)N_TOK * D_MODEL * 2);
    bf16_t* win_bf  = (bf16_t*)alloc((size_t)2 * D_INNER * D_MODEL * 2);
    bf16_t* wxp_bf  = (bf16_t*)alloc((size_t)XP_COLS * D_INNER * 2);
    bf16_t* wdt_bf  = (bf16_t*)alloc((size_t)D_INNER * DT_RANK * 2);
    bf16_t* wout_bf = (bf16_t*)alloc((size_t)D_MODEL * D_INNER * 2);
    float*  xz      = (float*) alloc((size_t)N_TOK * 2 * D_INNER * 4);
    float*  xconv_f = (float*) alloc((size_t)N_TOK * D_INNER * 4);
    bf16_t* xconv_b = (bf16_t*)alloc((size_t)N_TOK * D_INNER * 2);
    float*  xp      = (float*) alloc((size_t)N_TOK * XP_COLS * 4);
    bf16_t* dtlow_b = (bf16_t*)alloc((size_t)N_TOK * DT_RANK * 2);
    float*  dt      = (float*) alloc((size_t)N_TOK * D_INNER * 4);
    bf16_t* y_bf    = (bf16_t*)alloc((size_t)N_TOK * D_INNER * 2);

    const int CT = 256;
    auto cgrid = [&](size_t n) { return (unsigned)((n + CT - 1) / CT); };

    // 1) fp32 -> bf16 conversions
    k_f32_to_bf16<<<cgrid((size_t)N_TOK * D_MODEL), CT, 0, stream>>>(x, x_bf, N_TOK * D_MODEL);
    k_f32_to_bf16<<<cgrid((size_t)2 * D_INNER * D_MODEL), CT, 0, stream>>>(in_proj_w, win_bf, 2 * D_INNER * D_MODEL);
    k_f32_to_bf16<<<cgrid((size_t)XP_COLS * D_INNER), CT, 0, stream>>>(x_proj_w, wxp_bf, XP_COLS * D_INNER);
    k_f32_to_bf16<<<cgrid((size_t)D_INNER * DT_RANK), CT, 0, stream>>>(dt_proj_w, wdt_bf, D_INNER * DT_RANK);
    k_f32_to_bf16<<<cgrid((size_t)D_MODEL * D_INNER), CT, 0, stream>>>(out_proj_w, wout_bf, D_MODEL * D_INNER);

    // 2) xz = x · in_proj_w^T   (4096 x 4096 x 1024), 64x32 wave tiles
    k_gemm_bf16_wmma<4, 2><<<dim3(N_TOK / 64, (2 * D_INNER) / 32), 32, 0, stream>>>(
        x_bf, win_bf, xz, N_TOK, 2 * D_INNER, D_MODEL);

    // 3) causal depthwise conv + bias + SiLU
    k_conv_silu<<<cgrid((size_t)N_TOK * D_INNER), CT, 0, stream>>>(xz, conv_w, conv_b, xconv_f, xconv_b);

    // 4) xp = x_conv · x_proj_w^T   (4096 x 96 x 2048), 64x32 wave tiles
    k_gemm_bf16_wmma<4, 2><<<dim3(N_TOK / 64, XP_COLS / 32), 32, 0, stream>>>(
        xconv_b, wxp_bf, xp, N_TOK, XP_COLS, D_INNER);

    // 5) dt = softplus(dt_low · dt_proj_w^T + b)   (4096 x 2048 x 64)
    k_slice_to_bf16<<<cgrid((size_t)N_TOK * DT_RANK), CT, 0, stream>>>(xp, XP_COLS, 0, DT_RANK, dtlow_b, N_TOK);
    k_gemm_bf16_wmma<4, 2><<<dim3(N_TOK / 64, D_INNER / 32), 32, 0, stream>>>(
        dtlow_b, wdt_bf, dt, N_TOK, D_INNER, DT_RANK);
    k_bias_softplus<<<cgrid((size_t)N_TOK * D_INNER), CT, 0, stream>>>(dt, dt_proj_b, N_TOK, D_INNER);

    // 6) selective scan (+ D*x_conv, * silu(z)) -> y_bf
    k_scan<<<B_SZ * (D_INNER / 256), 256, 0, stream>>>(dt, xconv_f, xp, xz, A_log, D_param, y_bf);

    // 7) out = y · out_proj_w^T   (4096 x 1024 x 2048), 64x32 wave tiles
    k_gemm_bf16_wmma<4, 2><<<dim3(N_TOK / 64, D_MODEL / 32), 32, 0, stream>>>(
        y_bf, wout_bf, out, N_TOK, D_MODEL, D_INNER);
}
